// FCVT_43628277793391
// MI455X (gfx1250) — matrix operand order repaired and checked by reading, not compiled, not run on hardware
//
#include <hip/hip_runtime.h>
#include <hip/hip_bf16.h>

// ---------------------------------------------------------------------------
// FCVT (linear-attention ViT-Base) forward for gfx1250 (MI455X).
// Memory-bound (~1.3 TFLOP vs ~3-4 GB HBM traffic @23.3TB/s): all GEMMs run
// through one bf16 WMMA kernel (v_wmma_f32_16x16x32_bf16, f32 accumulate)
// that converts fp32 HBM data to bf16 in-register during LDS staging.
// Tokens padded 210 -> 224 per image so all GEMM dims are multiples of 32.
// GEMM variants are compile-time specialized (epilogue / B-orientation /
// bounds-guarding) so the hot path has zero per-element control flow.
// ---------------------------------------------------------------------------

#define NTOT   7168     // 32 * 224 padded tokens
#define TOKPAD 224
#define TOKVAL 210
#define CDIM   768
#define MDIM   3072
#define LAYERS 12
#define NCLS   1000

typedef __attribute__((ext_vector_type(16))) __bf16 v16bf;
typedef __attribute__((ext_vector_type(8)))  __bf16 v8bf;
typedef __attribute__((ext_vector_type(4)))  __bf16 v4bf;
typedef __attribute__((ext_vector_type(8)))  float  v8f;

union FragBf { v16bf v; v8bf h[2]; };

__device__ __forceinline__ __bf16 cvt_bf16(float f) {
  union { float f; unsigned int u; } a; a.f = f;
  unsigned int u = a.u;
  u += 0x7FFFu + ((u >> 16) & 1u);                    // round-to-nearest-even
  union { unsigned short s; __bf16 b; } o; o.s = (unsigned short)(u >> 16);
  return o.b;
}

__device__ __forceinline__ v4bf cvt4(float x, float y, float z, float w) {
  v4bf p; p[0] = cvt_bf16(x); p[1] = cvt_bf16(y); p[2] = cvt_bf16(z); p[3] = cvt_bf16(w);
  return p;
}

// ---------------------------------------------------------------------------
// Generic WMMA GEMM: D[m,n] = epi( sum_k A[m,k]*B[k,n] )
//  A: fp32, A[m,k] at A + m*lda + k                 (weights / qv)
//  B: fp32, BTRANS=0: B[k,n] at B + k*ldb + n       (activations, C-major)
//           BTRANS=1: B[k,n] at B + n*ldb + k       (for qv = pq * pv^T)
//  EPI: 0 none | 1 phi=elu+1 | 2 bias+gelu | 3 bias+residual
//  PAD (=EPI!=0): zero columns whose (n % 224) >= 210
//  GUARD: bounds-check M/N edges (only needed for attn, Nq=224)
//  K must be a multiple of 32 (true for 224/768/3072).
//  grid.z = batch with per-operand batch strides.
// Block tile 128x128, k-step 32, 8 waves (4Mx2N), each wave 32x64 (2x4 WMMA).
// ---------------------------------------------------------------------------
template <int EPI, bool BTRANS, bool GUARD>
__global__ void __launch_bounds__(256)
wmma_gemm(const float* __restrict__ A, int lda, long long batchA,
          const float* __restrict__ B, int ldb, long long batchB,
          float* __restrict__ D, int ldd, long long batchD,
          const float* __restrict__ bias,
          const float* __restrict__ resid,
          int Mq, int Nq, int K)
{
  constexpr bool PAD = (EPI != 0);
  constexpr int BM = 128, BN = 128, BK = 32, LDT = BK + 8;  // LDS row pad
  __shared__ __bf16 As[BM * LDT];
  __shared__ __bf16 Bs[BN * LDT];

  const long long bz = blockIdx.z;
  A += bz * batchA;  B += bz * batchB;  D += bz * batchD;
  const float* R = (EPI == 3) ? resid + bz * batchD : nullptr;

  const int t    = threadIdx.x;
  const int lane = t & 31;
  const int wave = t >> 5;
  const int wm   = (wave & 3) * 32;   // wave M origin within block tile
  const int wn   = (wave >> 2) * 64;  // wave N origin within block tile
  const int m0   = blockIdx.y * BM;
  const int n0   = blockIdx.x * BN;

  v8f acc[2][4];
  #pragma unroll
  for (int i = 0; i < 2; ++i)
    #pragma unroll
    for (int j = 0; j < 4; ++j)
      #pragma unroll
      for (int e = 0; e < 8; ++e) acc[i][j][e] = 0.0f;

  for (int k0 = 0; k0 < K; k0 += BK) {
    __syncthreads();
    // ---- stage A tile (fp32 -> bf16), LDS layout As[m][k]
    #pragma unroll
    for (int it = 0; it < 4; ++it) {
      int lin = it * 1024 + t * 4;
      int r = lin >> 5, c = lin & 31;
      const float* ap = A + (long long)(m0 + r) * lda + (k0 + c);
      float4 f;
      if (GUARD) {
        f = (m0 + r < Mq) ? *(const float4*)ap : float4{0.f, 0.f, 0.f, 0.f};
      } else {
        f = *(const float4*)ap;
        __builtin_prefetch(ap + BK, 0, 1);   // speculative global_prefetch_b8
      }
      *(v4bf*)&As[r * LDT + c] = cvt4(f.x, f.y, f.z, f.w);
    }
    // ---- stage B tile, LDS layout Bs[n][k] (fragment-friendly)
    if (BTRANS) {
      // B[k][n] at B + n*ldb + k : k contiguous, copy rows directly
      #pragma unroll
      for (int it = 0; it < 4; ++it) {
        int lin = it * 1024 + t * 4;
        int r = lin >> 5, c = lin & 31;   // r=n, c=k
        const float* bp = B + (long long)(n0 + r) * ldb + (k0 + c);
        float4 f;
        if (GUARD) {
          f = (n0 + r < Nq) ? *(const float4*)bp : float4{0.f, 0.f, 0.f, 0.f};
        } else {
          f = *(const float4*)bp;
          __builtin_prefetch(bp + BK, 0, 1);
        }
        *(v4bf*)&Bs[r * LDT + c] = cvt4(f.x, f.y, f.z, f.w);
      }
    } else {
      // B[k][n] n-contiguous: coalesced read along n, transpose into LDS
      #pragma unroll
      for (int it = 0; it < 4; ++it) {
        int lin = it * 1024 + t * 4;
        int kk = lin >> 7, nn = lin & 127;
        const float* bp = B + (long long)(k0 + kk) * ldb + (n0 + nn);
        float4 f;
        if (GUARD) {
          if (n0 + nn + 3 < Nq) {
            f = *(const float4*)bp;
          } else {
            f.x = (n0 + nn + 0 < Nq) ? bp[0] : 0.f;
            f.y = (n0 + nn + 1 < Nq) ? bp[1] : 0.f;
            f.z = (n0 + nn + 2 < Nq) ? bp[2] : 0.f;
            f.w = (n0 + nn + 3 < Nq) ? bp[3] : 0.f;
          }
        } else {
          f = *(const float4*)bp;
          __builtin_prefetch(bp + (long long)BK * ldb, 0, 1);
        }
        Bs[(nn + 0) * LDT + kk] = cvt_bf16(f.x);
        Bs[(nn + 1) * LDT + kk] = cvt_bf16(f.y);
        Bs[(nn + 2) * LDT + kk] = cvt_bf16(f.z);
        Bs[(nn + 3) * LDT + kk] = cvt_bf16(f.w);
      }
    }
    __syncthreads();

    // ---- fragment gathers per CDNA5 VGPR layouts, then 8 WMMAs
    const int half  = lane >> 4;     // lane half selects K sub-ranges
    const int idx16 = lane & 15;
    FragBf af[2], bfr[4];
    #pragma unroll
    for (int tm = 0; tm < 2; ++tm) {
      // A 16x32: lanes 0-15 -> K {0..7,16..23}; lanes 16-31 -> K {8..15,24..31}
      const __bf16* p = &As[(wm + tm * 16 + idx16) * LDT + half * 8];
      af[tm].h[0] = *(const v8bf*)(p);
      af[tm].h[1] = *(const v8bf*)(p + 16);
    }
    #pragma unroll
    for (int tn = 0; tn < 4; ++tn) {
      // B 32x16: lanes 0-15 -> K {0..7,16..23}; lanes 16-31 -> K {8..15,24..31}
      const __bf16* p = &Bs[(wn + tn * 16 + idx16) * LDT + half * 8];
      bfr[tn].h[0] = *(const v8bf*)(p);
      bfr[tn].h[1] = *(const v8bf*)(p + 16);
    }
    #pragma unroll
    for (int tm = 0; tm < 2; ++tm)
      #pragma unroll
      for (int tn = 0; tn < 4; ++tn)
        acc[tm][tn] = __builtin_amdgcn_wmma_f32_16x16x32_bf16(
            false, af[tm].v, false, bfr[tn].v, (short)0, acc[tm][tn], false, false);
  }

  // ---- fused epilogue (C/D layout: n = lane&15, m = base + e + 8*lanehalf)
  const int half  = lane >> 4;
  const int idx16 = lane & 15;
  #pragma unroll
  for (int tm = 0; tm < 2; ++tm) {
    #pragma unroll
    for (int tn = 0; tn < 4; ++tn) {
      int nn = n0 + wn + tn * 16 + idx16;
      if (GUARD && nn >= Nq) continue;
      bool padz = PAD && ((nn % TOKPAD) >= TOKVAL);
      int mbase = m0 + wm + tm * 16 + half * 8;
      #pragma unroll
      for (int e = 0; e < 8; ++e) {
        int mm = mbase + e;
        if (GUARD && mm >= Mq) continue;
        float v = acc[tm][tn][e];
        if (EPI == 1) {                       // phi = elu(x)+1
          v = (v > 0.f) ? (v + 1.f) : __expf(v);
        } else if (EPI == 2) {                // bias + exact GELU
          v += bias[mm];
          v = 0.5f * v * (1.f + erff(v * 0.70710678118654752f));
        } else if (EPI == 3) {                // bias + residual
          v += bias[mm] + R[(long long)mm * ldd + nn];
        }
        if (padz) v = 0.f;
        D[(long long)mm * ldd + nn] = v;
      }
    }
  }
}

// ---------------------------------------------------------------------------
// Channel LayerNorm over [C][NTOT]: one thread per token (coalesced across n).
// ---------------------------------------------------------------------------
__global__ void __launch_bounds__(256)
layernorm_ch(const float* __restrict__ h, const float* __restrict__ g,
             const float* __restrict__ b, float* __restrict__ out)
{
  int n = blockIdx.x * 256 + threadIdx.x;
  if (n >= NTOT) return;
  float s = 0.f, s2 = 0.f;
  for (int c = 0; c < CDIM; ++c) {
    float v = h[(long long)c * NTOT + n];
    s += v; s2 += v * v;
  }
  float mu  = s * (1.f / CDIM);
  float var = s2 * (1.f / CDIM) - mu * mu;
  var = var < 0.f ? 0.f : var;
  float rstd = rsqrtf(var + 1e-5f);
  for (int c = 0; c < CDIM; ++c) {
    float v = h[(long long)c * NTOT + n];
    out[(long long)c * NTOT + n] = (v - mu) * rstd * g[c] + b[c];
  }
}

// ---------------------------------------------------------------------------
// Patch embed + channel-LN + cls-token column + positional encoding.
// One block per (b, hh, ww); ww==0 is the cls column.
// ---------------------------------------------------------------------------
__global__ void __launch_bounds__(256)
patch_embed(const float* __restrict__ x,  const float* __restrict__ pw,
            const float* __restrict__ pb, const float* __restrict__ peg,
            const float* __restrict__ peb, const float* __restrict__ cls,
            const float* __restrict__ pe, float* __restrict__ h)
{
  const int b = blockIdx.x, hh = blockIdx.y, ww = blockIdx.z;  // ww in 0..14
  const int t = threadIdx.x;
  const int n = b * TOKPAD + hh * 15 + ww;
  if (ww == 0) {
    for (int c = t; c < CDIM; c += 256)
      h[(long long)c * NTOT + n] = cls[c] + pe[c * 210 + hh * 15];
    return;
  }
  __shared__ float patch[768];
  __shared__ float red[256];
  const int pc = ww - 1;
  for (int e = t; e < 768; e += 256) {
    int cc = e >> 8, rem = e & 255, i = rem >> 4, j = rem & 15;
    patch[e] = x[(((long long)b * 3 + cc) * 224 + hh * 16 + i) * 224 + pc * 16 + j];
  }
  __syncthreads();
  float ov[3];
  #pragma unroll
  for (int r = 0; r < 3; ++r) {
    int d = t + r * 256;
    const float* wrow = pw + (long long)d * 768;
    float acc = 0.f;
    for (int k = 0; k < 768; ++k) acc += wrow[k] * patch[k];
    ov[r] = acc + pb[d];
  }
  red[t] = ov[0] + ov[1] + ov[2];
  __syncthreads();
  for (int o = 128; o > 0; o >>= 1) { if (t < o) red[t] += red[t + o]; __syncthreads(); }
  float mu = red[0] * (1.f / 768.f);
  __syncthreads();
  float d0 = ov[0]-mu, d1 = ov[1]-mu, d2 = ov[2]-mu;
  red[t] = d0*d0 + d1*d1 + d2*d2;
  __syncthreads();
  for (int o = 128; o > 0; o >>= 1) { if (t < o) red[t] += red[t + o]; __syncthreads(); }
  float rstd = rsqrtf(red[0] * (1.f / 768.f) + 1e-5f);
  #pragma unroll
  for (int r = 0; r < 3; ++r) {
    int d = t + r * 256;
    h[(long long)d * NTOT + n] =
        (ov[r] - mu) * rstd * peg[d] + peb[d] + pe[d * 210 + hh * 15 + ww];
  }
}

__global__ void __launch_bounds__(256)
zero_f32(float* __restrict__ p, long long n)
{
  long long i = (long long)blockIdx.x * 256 + threadIdx.x;
  if (i < n) p[i] = 0.f;
}

__global__ void __launch_bounds__(256)
cls_pool(const float* __restrict__ h, float* __restrict__ cf)
{
  int idx = blockIdx.x * 256 + threadIdx.x;          // over 32*CDIM
  if (idx >= 32 * CDIM) return;
  int b = idx / CDIM, c = idx % CDIM;
  float s = 0.f;
  #pragma unroll
  for (int r = 0; r < 14; ++r) s += h[(long long)c * NTOT + b * TOKPAD + r * 15];
  cf[idx] = s * (1.f / 14.f);
}

__global__ void __launch_bounds__(256)
head_gemm(const float* __restrict__ cf, const float* __restrict__ hw,
          const float* __restrict__ hb, float* __restrict__ out)
{
  int idx = blockIdx.x * 256 + threadIdx.x;          // over 32*NCLS
  if (idx >= 32 * NCLS) return;
  int b = idx / NCLS, cl = idx % NCLS;
  const float* f  = cf + (long long)b * CDIM;
  const float* wr = hw + (long long)cl * CDIM;
  float acc = 0.f;
  for (int c = 0; c < CDIM; ++c) acc += f[c] * wr[c];
  out[idx] = acc + hb[cl];
}

// ---------------------------------------------------------------------------
extern "C" void kernel_launch(void* const* d_in, const int* in_sizes, int n_in,
                              void* d_out, int out_size, void* d_ws, size_t ws_size,
                              hipStream_t stream)
{
  (void)in_sizes; (void)n_in; (void)out_size; (void)ws_size;
  const float* x       = (const float*)d_in[0];
  const float* patch_w = (const float*)d_in[1];
  const float* patch_b = (const float*)d_in[2];
  const float* pe_g    = (const float*)d_in[3];
  const float* pe_b    = (const float*)d_in[4];
  const float* cls_tok = (const float*)d_in[5];
  const float* pos_enc = (const float*)d_in[6];
  const float* Wq      = (const float*)d_in[7];
  const float* Wk      = (const float*)d_in[8];
  const float* Wv      = (const float*)d_in[9];
  const float* Wo_w    = (const float*)d_in[10];
  const float* Wo_b    = (const float*)d_in[11];
  const float* ln1_g   = (const float*)d_in[12];
  const float* ln1_b   = (const float*)d_in[13];
  const float* ln2_g   = (const float*)d_in[14];
  const float* ln2_b   = (const float*)d_in[15];
  const float* m1w     = (const float*)d_in[16];
  const float* m1b     = (const float*)d_in[17];
  const float* m2w     = (const float*)d_in[18];
  const float* m2b     = (const float*)d_in[19];
  const float* hw      = (const float*)d_in[20];
  const float* hb      = (const float*)d_in[21];
  float* out = (float*)d_out;

  char* w = (char*)d_ws;
  auto alloc = [&](size_t elems) { float* p = (float*)w; w += elems * sizeof(float); return p; };
  float* h     = alloc((size_t)CDIM * NTOT);       // residual stream [C][NTOT]
  float* xn    = alloc((size_t)CDIM * NTOT);       // LN output
  float* pq    = alloc((size_t)CDIM * NTOT);       // phi(Wq xn); reused for attn
  float* pk    = alloc((size_t)CDIM * NTOT);
  float* pv    = alloc((size_t)CDIM * NTOT);
  float* qv    = alloc((size_t)32 * CDIM * CDIM);  // per-batch C x C
  float* tbuf  = alloc((size_t)MDIM * NTOT);       // MLP hidden
  float* cfeat = alloc((size_t)32 * CDIM);

  dim3 blk(256);
  { // zero h so the 14 pad columns per image stay zero
    long long tot = (long long)CDIM * NTOT;
    zero_f32<<<dim3((unsigned)((tot + 255) / 256)), blk, 0, stream>>>(h, tot);
  }
  patch_embed<<<dim3(32, 14, 15), blk, 0, stream>>>(x, patch_w, patch_b, pe_g, pe_b,
                                                    cls_tok, pos_enc, h);

  dim3 gLN((NTOT + 255) / 256);
  dim3 gP(NTOT / 128, CDIM / 128, 1);             // 56 x 6   full-token GEMMs
  dim3 gQV(CDIM / 128, CDIM / 128, 32);           // 6 x 6 x 32 batched qv
  dim3 gAT((TOKPAD + 127) / 128, CDIM / 128, 32); // 2 x 6 x 32 batched attn
  dim3 gM1(NTOT / 128, MDIM / 128, 1);            // 56 x 24  MLP up-proj

  for (int l = 0; l < LAYERS; ++l) {
    const float* wq = Wq   + (size_t)l * CDIM * CDIM;
    const float* wk = Wk   + (size_t)l * CDIM * CDIM;
    const float* wv = Wv   + (size_t)l * CDIM * CDIM;
    const float* wo = Wo_w + (size_t)l * CDIM * CDIM;
    const float* w1 = m1w  + (size_t)l * MDIM * CDIM;
    const float* w2 = m2w  + (size_t)l * CDIM * MDIM;

    layernorm_ch<<<gLN, blk, 0, stream>>>(h, ln1_g + l * CDIM, ln1_b + l * CDIM, xn);
    // pq/pk/pv = phi(W * xn), pad columns zeroed (phi(0)=1 must not leak)
    wmma_gemm<1, false, false><<<gP, blk, 0, stream>>>(
        wq, CDIM, 0, xn, NTOT, 0, pq, NTOT, 0, nullptr, nullptr, CDIM, NTOT, CDIM);
    wmma_gemm<1, false, false><<<gP, blk, 0, stream>>>(
        wk, CDIM, 0, xn, NTOT, 0, pk, NTOT, 0, nullptr, nullptr, CDIM, NTOT, CDIM);
    wmma_gemm<1, false, false><<<gP, blk, 0, stream>>>(
        wv, CDIM, 0, xn, NTOT, 0, pv, NTOT, 0, nullptr, nullptr, CDIM, NTOT, CDIM);
    // qv[b] = pq_b (C x 224) * pv_b^T -> C x C   (B transposed layout)
    wmma_gemm<0, true, false><<<gQV, blk, 0, stream>>>(
        pq, NTOT, TOKPAD, pv, NTOT, TOKPAD,
        qv, CDIM, (long long)CDIM * CDIM, nullptr, nullptr, CDIM, CDIM, TOKPAD);
    // attn[b] = qv[b] (C x C) * pk_b (C x 224) -> overwrite pq buffer
    wmma_gemm<0, false, true><<<gAT, blk, 0, stream>>>(
        qv, CDIM, (long long)CDIM * CDIM, pk, NTOT, TOKPAD,
        pq, NTOT, TOKPAD, nullptr, nullptr, CDIM, TOKPAD, CDIM);
    // h += Wo * attn + bias
    wmma_gemm<3, false, false><<<gP, blk, 0, stream>>>(
        wo, CDIM, 0, pq, NTOT, 0, h, NTOT, 0, Wo_b + l * CDIM, h, CDIM, NTOT, CDIM);
    layernorm_ch<<<gLN, blk, 0, stream>>>(h, ln2_g + l * CDIM, ln2_b + l * CDIM, xn);
    // t = gelu(W1 * xn + b1)
    wmma_gemm<2, false, false><<<gM1, blk, 0, stream>>>(
        w1, CDIM, 0, xn, NTOT, 0, tbuf, NTOT, 0, m1b + l * MDIM, nullptr, MDIM, NTOT, CDIM);
    // h += W2 * t + b2
    wmma_gemm<3, false, false><<<gP, blk, 0, stream>>>(
        w2, MDIM, 0, tbuf, NTOT, 0, h, NTOT, 0, m2b + l * CDIM, h, CDIM, NTOT, MDIM);
  }

  cls_pool<<<dim3((32 * CDIM + 255) / 256), blk, 0, stream>>>(h, cfeat);
  head_gemm<<<dim3((32 * NCLS + 255) / 256), blk, 0, stream>>>(cfeat, hw, hb, out);
}
